// BahdanauAttnEncoderDecoder_31997506355542
// MI455X (gfx1250) — matrix operand order repaired and checked by reading, not compiled, not run on hardware
//
#include <hip/hip_runtime.h>
#include <hip/hip_bf16.h>

// ---------------- problem constants ----------------
constexpr int B     = 32;
constexpr int S_ENC = 400;
constexpr int S_DEC = 64;
constexpr int E     = 256;
constexpr int H     = 512;
constexpr int V     = 32000;
constexpr int G     = 4 * H;        // 2048 gate width

// ---------------- types ----------------
typedef __attribute__((ext_vector_type(16))) __bf16 bf16x16;
typedef __attribute__((ext_vector_type(8)))  float  f32x8;

union Frag { uint4 u[2]; bf16x16 v; };

__device__ __forceinline__ unsigned short f2bf(float f) {
  unsigned int u = __float_as_uint(f);
  unsigned int r = (u + 0x7FFFu + ((u >> 16) & 1u)) >> 16;
  return (unsigned short)r;
}
__device__ __forceinline__ float sigf(float x) { return 1.f / (1.f + expf(-x)); }

// ---------------------------------------------------------------------------
// WMMA GEMM:  C[M x N] = A[M x K] * W[N x K]^T (+ Add)(+ bias)
// A, W bf16 row-major; C fp32 row stride ldc. M must be a multiple of 32.
// Each wave computes a 32x16 output block (two 16x16 accumulators) so the
// B (weight) fragment is loaded once and reused by two WMMAs per k-step.
// ---------------------------------------------------------------------------
__global__ void wmma_gemm_xwT(const unsigned short* __restrict__ A, int lda,
                              const unsigned short* __restrict__ W, int ldw,
                              float* __restrict__ C, long long ldc,
                              const float* __restrict__ Add, long long ldadd,
                              const float* __restrict__ bias,
                              int mGroups, int tilesN, int K)
{
  int wave  = (blockIdx.x * blockDim.x + threadIdx.x) >> 5;
  if (wave >= mGroups * tilesN) return;         // wave-uniform exit
  int mg    = wave / tilesN;                    // rows mg*32 .. mg*32+31
  int tn    = wave % tilesN;
  int lane  = threadIdx.x & 31;
  int half  = lane >> 4;                        // 0: lanes 0-15, 1: lanes 16-31
  int l16   = lane & 15;

  const unsigned short* arow0 = A + (size_t)(mg * 32 + l16) * lda;
  const unsigned short* arow1 = arow0 + (size_t)16 * lda;
  const unsigned short* wrow  = W + (size_t)(tn * 16 + l16) * ldw;

  f32x8 acc0 = {0.f, 0.f, 0.f, 0.f, 0.f, 0.f, 0.f, 0.f};
  f32x8 acc1 = {0.f, 0.f, 0.f, 0.f, 0.f, 0.f, 0.f, 0.f};

  for (int k0 = 0; k0 < K; k0 += 32) {
    // prefetch next K-chunk of the weight stream (global_prefetch_b8)
    __builtin_prefetch(wrow + k0 + 64, 0, 1);
    Frag a0, a1, b;
    // B 32x16 fragment: lane's column = l16; lanes 0-15 hold K[0..7],[16..23]; lanes 16-31 K[8..15],[24..31]
    b.u[0]  = *reinterpret_cast<const uint4*>(wrow + k0 + half * 8);
    b.u[1]  = *reinterpret_cast<const uint4*>(wrow + k0 + 16 + half * 8);
    // A 16x32 fragments: lanes 0-15 hold K[0..7],[16..23]; lanes 16-31 K[8..15],[24..31]
    a0.u[0] = *reinterpret_cast<const uint4*>(arow0 + k0 + half * 8);
    a0.u[1] = *reinterpret_cast<const uint4*>(arow0 + k0 + 16 + half * 8);
    a1.u[0] = *reinterpret_cast<const uint4*>(arow1 + k0 + half * 8);
    a1.u[1] = *reinterpret_cast<const uint4*>(arow1 + k0 + 16 + half * 8);
    acc0 = __builtin_amdgcn_wmma_f32_16x16x32_bf16(false, a0.v, false, b.v,
                                                   (short)0, acc0, false, false);
    acc1 = __builtin_amdgcn_wmma_f32_16x16x32_bf16(false, a1.v, false, b.v,
                                                   (short)0, acc1, false, false);
  }

  int col = tn * 16 + l16;
  float bv = bias ? bias[col] : 0.f;
#pragma unroll
  for (int r = 0; r < 8; ++r) {
    int row0 = mg * 32 + half * 8 + r;          // C/D layout: VGPR r -> row r + 8*half
    int row1 = row0 + 16;
    float v0 = acc0[r] + bv;
    float v1 = acc1[r] + bv;
    if (Add) {
      v0 += Add[(long long)row0 * ldadd + col];
      v1 += Add[(long long)row1 * ldadd + col];
    }
    C[(long long)row0 * ldc + col] = v0;
    C[(long long)row1 * ldc + col] = v1;
  }
}

// ---------------- elementwise helpers ----------------
__global__ void cvt_bf16_kernel(const float* __restrict__ s, unsigned short* __restrict__ d,
                                long long n) {
  long long i = (long long)blockIdx.x * blockDim.x + threadIdx.x;
  if (i < n) d[i] = f2bf(s[i]);
}

__global__ void bias_sum_kernel(const float* a, const float* b, float* d, int n) {
  int i = blockIdx.x * blockDim.x + threadIdx.x;
  if (i < n) d[i] = a[i] + b[i];
}

__global__ void embed_kernel(const int* __restrict__ tok, const float* __restrict__ emb,
                             unsigned short* __restrict__ out, long long ntok, int Ein) {
  long long i = (long long)blockIdx.x * blockDim.x + threadIdx.x;
  if (i >= ntok * Ein) return;
  long long t = i / Ein;
  int e = (int)(i - t * Ein);
  out[i] = f2bf(emb[(size_t)tok[t] * Ein + e]);
}

// ---------------- LSTM pointwise ----------------
__global__ void lstm_pointwise_kernel(const float* __restrict__ gates,
                                      float* __restrict__ c, float* __restrict__ h,
                                      unsigned short* __restrict__ hbf,
                                      float* __restrict__ seq,          // nullable: enc_out[.,t,.]
                                      unsigned short* __restrict__ seqbf,
                                      long long bstride, int Hh) {
  int idx = blockIdx.x * blockDim.x + threadIdx.x;     // B*H threads
  int b = idx / Hh, j = idx - b * Hh;
  if (b >= B) return;
  const float* g = gates + (size_t)b * 4 * Hh;
  float gi = sigf(g[j]);
  float gf = sigf(g[Hh + j]);
  float gg = tanhf(g[2 * Hh + j]);
  float go = sigf(g[3 * Hh + j]);
  float cn = gf * c[idx] + gi * gg;
  float hn = go * tanhf(cn);
  c[idx] = cn; h[idx] = hn; hbf[idx] = f2bf(hn);
  if (seq) {
    seq[(long long)b * bstride + j]   = hn;
    seqbf[(long long)b * bstride + j] = f2bf(hn);
  }
}

// ---------------- attention ----------------
__global__ void attn_scores_kernel(const float* __restrict__ enc_proj,
                                   const float* __restrict__ q,
                                   const float* __restrict__ v,
                                   float* __restrict__ scores, int S, int Hh, int total) {
  int gw = (blockIdx.x * blockDim.x + threadIdx.x) >> 5;    // one wave per (b,s)
  if (gw >= total) return;
  int b = gw / S, s = gw - b * S;
  int lane = threadIdx.x & 31;
  const float* ep = enc_proj + ((size_t)b * S + s) * Hh;
  const float* qb = q + (size_t)b * Hh;
  float acc = 0.f;
  for (int k = lane; k < Hh; k += 32)
    acc += v[k] * tanhf(ep[k] + qb[k]);
  for (int m = 16; m; m >>= 1) acc += __shfl_xor(acc, m, 32);
  if (lane == 0) scores[(size_t)b * S + s] = acc;
}

__global__ void softmax_kernel(const float* __restrict__ scores, float* __restrict__ attn, int S) {
  __shared__ float red[256];
  int b = blockIdx.x, t = threadIdx.x;
  const float* x = scores + (size_t)b * S;
  float m = -1e30f;
  for (int s = t; s < S; s += 256) m = fmaxf(m, x[s]);
  red[t] = m; __syncthreads();
  for (int o = 128; o; o >>= 1) { if (t < o) red[t] = fmaxf(red[t], red[t + o]); __syncthreads(); }
  m = red[0]; __syncthreads();
  float sum = 0.f;
  for (int s = t; s < S; s += 256) { float e = expf(x[s] - m); attn[(size_t)b * S + s] = e; sum += e; }
  red[t] = sum; __syncthreads();
  for (int o = 128; o; o >>= 1) { if (t < o) red[t] += red[t + o]; __syncthreads(); }
  float inv = 1.f / red[0];
  for (int s = t; s < S; s += 256) attn[(size_t)b * S + s] *= inv;
}

__global__ void context_kernel(const float* __restrict__ attn,
                               const float* __restrict__ enc_out,
                               unsigned short* __restrict__ rnn_in,
                               int S, int Hh, int Ein) {
  int idx = blockIdx.x * blockDim.x + threadIdx.x;      // B*H threads
  int b = idx / Hh, h = idx - b * Hh;
  if (b >= B) return;
  const float* eo = enc_out + (size_t)b * S * Hh + h;
  const float* at = attn + (size_t)b * S;
  float acc = 0.f;
  for (int s = 0; s < S; ++s) acc += at[s] * eo[(size_t)s * Hh];
  rnn_in[(size_t)b * (Ein + Hh) + Ein + h] = f2bf(acc);
}

__global__ void copy_decx_kernel(const unsigned short* __restrict__ dec_x,
                                 unsigned short* __restrict__ rnn_in,
                                 int t, int Sdec, int Ein, int Hh) {
  int idx = blockIdx.x * blockDim.x + threadIdx.x;      // B*E threads
  int b = idx / Ein, e = idx - b * Ein;
  if (b >= B) return;
  rnn_in[(size_t)b * (Ein + Hh) + e] = dec_x[((size_t)b * Sdec + t) * Ein + e];
}

// ---------------- output finishing ----------------
__global__ void log_softmax_kernel(float* __restrict__ out, int Vv) {
  __shared__ float red[256];
  long long row = blockIdx.x;
  float* x = out + row * (long long)Vv;
  int t = threadIdx.x;
  float m = -1e30f;
  for (int i = t; i < Vv; i += 256) m = fmaxf(m, x[i]);
  red[t] = m; __syncthreads();
  for (int o = 128; o; o >>= 1) { if (t < o) red[t] = fmaxf(red[t], red[t + o]); __syncthreads(); }
  m = red[0]; __syncthreads();
  float sum = 0.f;
  for (int i = t; i < Vv; i += 256) sum += expf(x[i] - m);
  red[t] = sum; __syncthreads();
  for (int o = 128; o; o >>= 1) { if (t < o) red[t] += red[t + o]; __syncthreads(); }
  float lse = m + logf(red[0]);
  __syncthreads();
  for (int i = t; i < Vv; i += 256) x[i] = x[i] - lse;
}

__global__ void copy_hc_kernel(const float* h, const float* c, float* dst, int n) {
  int i = blockIdx.x * blockDim.x + threadIdx.x;
  if (i < n) { dst[i] = h[i]; dst[n + i] = c[i]; }
}

// ---------------- host side ----------------
static void launch_gemm(hipStream_t st, const unsigned short* A, int lda,
                        const unsigned short* W, int ldw,
                        float* C, long long ldc,
                        const float* Add, long long ldadd,
                        const float* bias, int M, int N, int K) {
  int mg = M >> 5;                 // 32 rows per wave
  int tn = N >> 4;
  long long waves = (long long)mg * tn;
  int blocks = (int)((waves * 32 + 255) / 256);
  wmma_gemm_xwT<<<blocks, 256, 0, st>>>(A, lda, W, ldw, C, ldc, Add, ldadd, bias, mg, tn, K);
}

extern "C" void kernel_launch(void* const* d_in, const int* in_sizes, int n_in,
                              void* d_out, int out_size, void* d_ws, size_t ws_size,
                              hipStream_t stream) {
  const int*   enc_in  = (const int*)d_in[0];
  const int*   dec_in  = (const int*)d_in[1];
  const float* enc_h0  = (const float*)d_in[2];
  const float* enc_c0  = (const float*)d_in[3];
  const float* enc_emb = (const float*)d_in[4];
  const float* dec_emb = (const float*)d_in[5];
  const float* enc_Wih = (const float*)d_in[6];
  const float* enc_Whh = (const float*)d_in[7];
  const float* enc_bih = (const float*)d_in[8];
  const float* enc_bhh = (const float*)d_in[9];
  const float* attn_W  = (const float*)d_in[10];
  const float* attn_b  = (const float*)d_in[11];
  const float* attn_v  = (const float*)d_in[12];
  const float* dec_Wih = (const float*)d_in[13];
  const float* dec_Whh = (const float*)d_in[14];
  const float* dec_bih = (const float*)d_in[15];
  const float* dec_bhh = (const float*)d_in[16];
  const float* fc_W    = (const float*)d_in[17];
  const float* fc_b    = (const float*)d_in[18];
  float* out = (float*)d_out;

  // workspace carve-up
  char* base = (char*)d_ws;
  size_t off = 0;
  auto alloc = [&](size_t bytes) -> char* {
    char* p = base + off;
    off = (off + bytes + 255) & ~(size_t)255;
    return p;
  };
  unsigned short* encx    = (unsigned short*)alloc((size_t)B * S_ENC * E * 2);
  unsigned short* decx    = (unsigned short*)alloc((size_t)B * S_DEC * E * 2);
  float*          Gx      = (float*)        alloc((size_t)B * S_ENC * G * 4);
  float*          encout  = (float*)        alloc((size_t)B * S_ENC * H * 4);
  unsigned short* encoutb = (unsigned short*)alloc((size_t)B * S_ENC * H * 2);
  float*          encproj = (float*)        alloc((size_t)B * S_ENC * H * 4);
  unsigned short* wih_bf  = (unsigned short*)alloc((size_t)G * E * 2);
  unsigned short* whh_bf  = (unsigned short*)alloc((size_t)G * H * 2);
  unsigned short* attnW_bf= (unsigned short*)alloc((size_t)H * 2 * H * 2);
  unsigned short* dwih_bf = (unsigned short*)alloc((size_t)G * (E + H) * 2);
  unsigned short* dwhh_bf = (unsigned short*)alloc((size_t)G * H * 2);
  unsigned short* fcw_bf  = (unsigned short*)alloc((size_t)V * H * 2);
  float*          encb    = (float*)        alloc((size_t)G * 4);
  float*          decb    = (float*)        alloc((size_t)G * 4);
  float*          hbuf    = (float*)        alloc((size_t)B * H * 4);
  float*          cbuf    = (float*)        alloc((size_t)B * H * 4);
  unsigned short* hbf     = (unsigned short*)alloc((size_t)B * H * 2);
  float*          gates   = (float*)        alloc((size_t)B * G * 4);
  float*          qbuf    = (float*)        alloc((size_t)B * H * 4);
  float*          scores  = (float*)        alloc((size_t)B * S_ENC * 4);
  float*          attnw   = (float*)        alloc((size_t)B * S_ENC * 4);
  unsigned short* rnnin   = (unsigned short*)alloc((size_t)B * (E + H) * 2);

  auto cvt = [&](const float* s, unsigned short* d, long long n) {
    cvt_bf16_kernel<<<(int)((n + 255) / 256), 256, 0, stream>>>(s, d, n);
  };

  // weights -> bf16, fused biases
  cvt(enc_Wih, wih_bf,  (long long)G * E);
  cvt(enc_Whh, whh_bf,  (long long)G * H);
  cvt(attn_W,  attnW_bf,(long long)H * 2 * H);
  cvt(dec_Wih, dwih_bf, (long long)G * (E + H));
  cvt(dec_Whh, dwhh_bf, (long long)G * H);
  cvt(fc_W,    fcw_bf,  (long long)V * H);
  bias_sum_kernel<<<(G + 255) / 256, 256, 0, stream>>>(enc_bih, enc_bhh, encb, G);
  bias_sum_kernel<<<(G + 255) / 256, 256, 0, stream>>>(dec_bih, dec_bhh, decb, G);

  // embeddings -> bf16
  {
    long long n1 = (long long)B * S_ENC * E;
    embed_kernel<<<(int)((n1 + 255) / 256), 256, 0, stream>>>(enc_in, enc_emb, encx, (long long)B * S_ENC, E);
    long long n2 = (long long)B * S_DEC * E;
    embed_kernel<<<(int)((n2 + 255) / 256), 256, 0, stream>>>(dec_in, dec_emb, decx, (long long)B * S_DEC, E);
  }

  // G_x = enc_x @ enc_Wih^T + (bih+bhh)   [12800 x 2048]
  launch_gemm(stream, encx, E, wih_bf, E, Gx, G, nullptr, 0, encb, B * S_ENC, G, E);

  // init h, c
  hipMemcpyAsync(hbuf, enc_h0, (size_t)B * H * 4, hipMemcpyDeviceToDevice, stream);
  hipMemcpyAsync(cbuf, enc_c0, (size_t)B * H * 4, hipMemcpyDeviceToDevice, stream);
  cvt(hbuf, hbf, (long long)B * H);

  // ----- encoder recurrence -----
  for (int t = 0; t < S_ENC; ++t) {
    launch_gemm(stream, hbf, H, whh_bf, H, gates, G,
                Gx + (long long)t * G, (long long)S_ENC * G, nullptr, B, G, H);
    lstm_pointwise_kernel<<<(B * H) / 256, 256, 0, stream>>>(
        gates, cbuf, hbuf, hbf,
        encout + (long long)t * H, encoutb + (long long)t * H, (long long)S_ENC * H, H);
  }

  // enc_proj = enc_out @ W_e^T   (W_e = attn_W[:, H:], row stride 2H)
  launch_gemm(stream, encoutb, H, attnW_bf + H, 2 * H, encproj, H,
              nullptr, 0, nullptr, B * S_ENC, H, H);

  // ----- decoder recurrence -----
  for (int t = 0; t < S_DEC; ++t) {
    // q = h @ W_q^T + attn_b
    launch_gemm(stream, hbf, H, attnW_bf, 2 * H, qbuf, H, nullptr, 0, attn_b, B, H, H);
    attn_scores_kernel<<<(B * S_ENC * 32) / 256, 256, 0, stream>>>(
        encproj, qbuf, attn_v, scores, S_ENC, H, B * S_ENC);
    softmax_kernel<<<B, 256, 0, stream>>>(scores, attnw, S_ENC);
    copy_decx_kernel<<<(B * E) / 256, 256, 0, stream>>>(decx, rnnin, t, S_DEC, E, H);
    context_kernel<<<(B * H) / 256, 256, 0, stream>>>(attnw, encout, rnnin, S_ENC, H, E);
    // gates = rnn_in @ dec_Wih^T + (bih+bhh) ;  gates += h @ dec_Whh^T
    launch_gemm(stream, rnnin, E + H, dwih_bf, E + H, gates, G, nullptr, 0, decb, B, G, E + H);
    launch_gemm(stream, hbf, H, dwhh_bf, H, gates, G, gates, G, nullptr, B, G, H);
    lstm_pointwise_kernel<<<(B * H) / 256, 256, 0, stream>>>(
        gates, cbuf, hbuf, hbf, nullptr, nullptr, 0, H);
    // logits -> d_out[:, t, :]  (row stride S_DEC*V)
    launch_gemm(stream, hbf, H, fcw_bf, H, out + (long long)t * V, (long long)S_DEC * V,
                nullptr, 0, fc_b, B, V, H);
  }

  // log_softmax over vocab, in place on d_out
  log_softmax_kernel<<<B * S_DEC, 256, 0, stream>>>(out, V);
  // final (h, c) appended after outputs
  copy_hc_kernel<<<(B * H + 255) / 256, 256, 0, stream>>>(
      hbuf, cbuf, out + (long long)B * S_DEC * V, B * H);
}
